// ANIModel_80032420593970
// MI455X (gfx1250) — matrix operand order repaired and checked by reading, not compiled, not run on hardware
//
#include <hip/hip_runtime.h>
#include <hip/hip_bf16.h>

typedef __attribute__((ext_vector_type(16))) __bf16 v16bf;
typedef __attribute__((ext_vector_type(8)))  __bf16 v8bf;
typedef __attribute__((ext_vector_type(8)))  float  v8f;

#define NSPEC    7
#define NB       1024
#define NA       48
#define NATOMS   (NB*NA)          // 49152
#define TILE_M   64               // atoms per block (4 WMMA M-subtiles per B-fragment)
#define MAXT     784              // ceil(49152/64) + per-species padding slack
#define PERM_PAD 49600            // 49152 + 7*64

// packed bf16 weight region (B-fragment order), per species:
#define L1_FRAGS (16*12)          // N-tiles * K-tiles for 384->256
#define L2_FRAGS (12*8)           //                      256->192
#define L3_FRAGS (10*6)           //                      192->160
#define SPEC_ELEMS ((L1_FRAGS+L2_FRAGS+L3_FRAGS)*512)   // 178176 bf16
#define WP_ELEMS   (NSPEC*SPEC_ELEMS)                   // 1247232 bf16
#define WP_BYTES   ((size_t)WP_ELEMS*2)                 // 2.49 MB (64B aligned)
#define L2_OFF     (L1_FRAGS*512)
#define L3_OFF     ((L1_FRAGS+L2_FRAGS)*512)

__device__ __forceinline__ float celu01(float x) {
    return x > 0.f ? x : 0.1f * (__expf(x * 10.f) - 1.f);
}

// ---------------- init: zero counters/cursors, -1 tiles/perm, zero out ----------------
__global__ void init_kernel(int* __restrict__ ip, int* __restrict__ tspec,
                            int* __restrict__ tbase, int* __restrict__ perm,
                            float* __restrict__ out) {
    int i = blockIdx.x * 256 + threadIdx.x;
    if (i < 16)       ip[i] = 0;          // counts[8] + cursor[8]
    if (i < MAXT)     { tspec[i] = -1; tbase[i] = -1; }
    if (i < PERM_PAD) perm[i] = -1;
    if (i < NB)       out[i] = 0.f;
}

// ---------------- pre-pack fp32 weights into bf16 WMMA B-fragment order ----------------
// B-frag layout (32x16 bf16, 16 elems/lane): elem e at lane L -> N = L%16, K = (L/16)*16 + e
__global__ void prepack_kernel(const float* __restrict__ W1,
                               const float* __restrict__ W2,
                               const float* __restrict__ W3,
                               __bf16* __restrict__ wp) {
    int i = blockIdx.x * 256 + threadIdx.x;
    if (i >= WP_ELEMS) return;
    int s = i / SPEC_ELEMS;
    int r = i - s * SPEC_ELEMS;
    const float* W; int KT, Kdim, r0;
    if (r < L2_OFF)      { W = W1 + s * 256 * 384; KT = 12; Kdim = 384; r0 = r; }
    else if (r < L3_OFF) { W = W2 + s * 192 * 256; KT = 8;  Kdim = 256; r0 = r - L2_OFF; }
    else                 { W = W3 + s * 160 * 192; KT = 6;  Kdim = 192; r0 = r - L3_OFF; }
    int f = r0 >> 9, q = r0 & 511;
    int lane = q >> 4, e = q & 15;
    int nt = f / KT,  kt = f - nt * KT;
    int k = kt * 32 + (lane >> 4) * 16 + e;
    int n = nt * 16 + (lane & 15);
    wp[i] = (__bf16)W[n * Kdim + k];   // B[k][n] = W[n][k]  (out = x @ W^T)
}

// ---------------- species histogram ----------------
__global__ void hist_kernel(const int* __restrict__ species, int* __restrict__ counts) {
    int i = blockIdx.x * 256 + threadIdx.x;
    if (i >= NATOMS) return;
    int s = species[i];
    if (s >= 0 && s < NSPEC) atomicAdd(&counts[s], 1);
}

// ---------------- scan: segment starts (64-padded) + tile descriptors ----------------
__global__ void scan_kernel(const int* __restrict__ counts, int* __restrict__ cursor,
                            int* __restrict__ tspec, int* __restrict__ tbase) {
    if (threadIdx.x != 0 || blockIdx.x != 0) return;
    int off = 0, t = 0;
    for (int s = 0; s < NSPEC; ++s) {
        int c = counts[s];
        cursor[s] = off;
        int nt = (c + TILE_M - 1) / TILE_M;
        for (int i = 0; i < nt; ++i) { tspec[t] = s; tbase[t] = off + i * TILE_M; ++t; }
        off += nt * TILE_M;
    }
}

// ---------------- scatter atoms into species-sorted order ----------------
__global__ void scatter_kernel(const int* __restrict__ species, int* __restrict__ cursor,
                               int* __restrict__ perm) {
    int i = blockIdx.x * 256 + threadIdx.x;
    if (i >= NATOMS) return;
    int s = species[i];
    if (s < 0 || s >= NSPEC) return;
    int p = atomicAdd(&cursor[s], 1);
    perm[p] = i;
}

// ------ one fused MLP layer on a 64-atom tile: each B-frag feeds FOUR wmma (M reuse) ----
// A-frag (16x32 bf16): lane<16 -> M=lane, elems 0..15 = K 0..15
//                      lane>=16 -> M=lane-16, elems 0..15 = K 16..31
template<int Kin, int Nout, int KT, int NT>
__device__ __forceinline__ void mlp_layer(const __bf16* sIn, __bf16* sOut,
                                          const __bf16* __restrict__ wfrag,
                                          const float* __restrict__ bias,
                                          int wave, int lane) {
    const int m    = lane & 15;
    const int half = lane >> 4;
    for (int nt = wave; nt < NT; nt += 4) {
        v8f acc[4] = {};
        for (int kt = 0; kt < KT; ++kt) {
            v16bf b = *(const v16bf*)(wfrag + ((size_t)(nt * KT + kt) << 9) + (lane << 4));
            #pragma unroll
            for (int mt = 0; mt < 4; ++mt) {
                const __bf16* pa = sIn + (mt * 16 + m) * Kin + kt * 32 + half * 16;
                v8bf lo = *(const v8bf*)pa;
                v8bf hi = *(const v8bf*)(pa + 8);
                v16bf a;
                #pragma unroll
                for (int e = 0; e < 8; ++e) { a[e] = lo[e]; a[8 + e] = hi[e]; }
                acc[mt] = __builtin_amdgcn_wmma_f32_16x16x32_bf16(
                              false, a, false, b, (short)0, acc[mt], false, false);
            }
        }
        const int n  = nt * 16 + m;
        const float bv = bias[n];
        #pragma unroll
        for (int mt = 0; mt < 4; ++mt) {
            #pragma unroll
            for (int r = 0; r < 8; ++r) {
                // D layout: M = r + 8*half, N = lane%16 ; subtile mt at rows +16*mt
                sOut[(mt * 16 + half * 8 + r) * Nout + n] = (__bf16)celu01(acc[mt][r] + bv);
            }
        }
    }
}

// ---------------- grouped GEMM: 1 block = 1 species-uniform 64-atom tile ----------------
__global__ void __launch_bounds__(128) gemm_kernel(
    const float* __restrict__ aev, const __bf16* __restrict__ wp,
    const float* __restrict__ b1, const float* __restrict__ b2,
    const float* __restrict__ b3, const float* __restrict__ W4,
    const float* __restrict__ b4,
    const int* __restrict__ tspec, const int* __restrict__ tbase,
    const int* __restrict__ perm, float* __restrict__ out) {

    // ping-pong LDS: aev->A, L1->B, L2->A, L3->B  (80 KB total; WGP has 320 KB)
    __shared__ __bf16 bufA[TILE_M * 384];   // 48 KB  (aev K=384 / h2 N=192)
    __shared__ __bf16 bufB[TILE_M * 256];   // 32 KB  (h1 N=256 / h3 N=160)
    __shared__ float  sRed[128];
    __shared__ int    sIdx[TILE_M];
    __shared__ int    sMeta[2];

    const int tid  = threadIdx.x;
    const int wave = tid >> 5;
    const int lane = tid & 31;

    if (tid == 0) { sMeta[0] = tspec[blockIdx.x]; sMeta[1] = tbase[blockIdx.x]; }
    __syncthreads();
    const int spec = sMeta[0];
    if (spec < 0) return;                      // whole block exits together (EXEC stays full)
    if (tid < TILE_M) sIdx[tid] = perm[sMeta[1] + tid];
    __syncthreads();

    // stage aev tile fp32 -> bf16 row-major in LDS (thread: atom tid/2, 192-float chunk)
    {
        const int a  = tid >> 1;
        const int c0 = (tid & 1) * 192;
        const int ai = sIdx[a];
        __bf16* dst = &bufA[a * 384 + c0];
        if (ai >= 0) {
            const float4* src = (const float4*)(aev + (size_t)ai * 384 + c0);
            #pragma unroll
            for (int j = 0; j < 48; ++j) {
                float4 v = src[j];
                dst[j*4+0] = (__bf16)v.x; dst[j*4+1] = (__bf16)v.y;
                dst[j*4+2] = (__bf16)v.z; dst[j*4+3] = (__bf16)v.w;
            }
        } else {
            #pragma unroll
            for (int j = 0; j < 192; ++j) dst[j] = (__bf16)0.f;
        }
    }
    __syncthreads();

    const __bf16* wps = wp + (size_t)spec * SPEC_ELEMS;
    mlp_layer<384, 256, 12, 16>(bufA, bufB, wps,          b1 + spec * 256, wave, lane);
    __syncthreads();
    mlp_layer<256, 192,  8, 12>(bufB, bufA, wps + L2_OFF, b2 + spec * 192, wave, lane);
    __syncthreads();
    mlp_layer<192, 160,  6, 10>(bufA, bufB, wps + L3_OFF, b3 + spec * 160, wave, lane);
    __syncthreads();

    // layer 4: 160 -> 1 dot per atom (2 threads x 80 elems), scatter-sum per molecule
    {
        const int a  = tid >> 1;
        const int c0 = (tid & 1) * 80;
        const float* w4 = W4 + spec * 160;
        float acc = 0.f;
        #pragma unroll 8
        for (int e = 0; e < 80; ++e)
            acc += (float)bufB[a * 160 + c0 + e] * w4[c0 + e];
        sRed[tid] = acc;
    }
    __syncthreads();
    if (tid < TILE_M) {
        const int ai = sIdx[tid];
        if (ai >= 0) {
            float v = sRed[tid*2] + sRed[tid*2+1] + b4[spec];
            atomicAdd(out + (ai / NA), v);
        }
    }
}

extern "C" void kernel_launch(void* const* d_in, const int* in_sizes, int n_in,
                              void* d_out, int out_size, void* d_ws, size_t ws_size,
                              hipStream_t stream) {
    const int*   species = (const int*)  d_in[0];
    const float* aev     = (const float*)d_in[1];
    const float* W1      = (const float*)d_in[2];
    const float* b1      = (const float*)d_in[3];
    const float* W2      = (const float*)d_in[4];
    const float* b2      = (const float*)d_in[5];
    const float* W3      = (const float*)d_in[6];
    const float* b3      = (const float*)d_in[7];
    const float* W4      = (const float*)d_in[8];
    const float* b4      = (const float*)d_in[9];
    float* out = (float*)d_out;

    __bf16* wp = (__bf16*)d_ws;                       // packed weights first (32B aligned)
    int* ip     = (int*)((char*)d_ws + WP_BYTES);
    int* counts = ip;
    int* cursor = ip + 8;
    int* tspec  = ip + 16;
    int* tbase  = tspec + MAXT;
    int* perm   = tbase + MAXT;
    size_t needed = WP_BYTES + (size_t)(16 + 2 * MAXT + PERM_PAD) * 4;
    if (ws_size < needed) return;

    init_kernel   <<<(PERM_PAD + 255) / 256, 256, 0, stream>>>(ip, tspec, tbase, perm, out);
    prepack_kernel<<<(WP_ELEMS + 255) / 256, 256, 0, stream>>>(W1, W2, W3, wp);
    hist_kernel   <<<(NATOMS   + 255) / 256, 256, 0, stream>>>(species, counts);
    scan_kernel   <<<1, 32, 0, stream>>>(counts, cursor, tspec, tbase);
    scatter_kernel<<<(NATOMS   + 255) / 256, 256, 0, stream>>>(species, cursor, perm);
    gemm_kernel   <<<MAXT, 128, 0, stream>>>(aev, wp, b1, b2, b3, W4, b4,
                                             tspec, tbase, perm, out);
}
